// LoRA_Attention_3633542332621
// MI455X (gfx1250) — matrix operand order repaired and checked by reading, not compiled, not run on hardware
//
#include <hip/hip_runtime.h>
#include <hip/hip_bf16.h>
#include <math.h>

// ---------------------------------------------------------------------------
// LoRA attention forward for MI455X (gfx1250), wave32 + WMMA bf16.
//   x[2,2048,1024] f32 -> out[2,2048,1024] f32
// All GEMMs run on v_wmma_f32_16x16x32_bf16 (f32 accumulate).
// Attention K/V tiles are double-buffered in LDS via global_load_async_to_lds
// (ASYNCcnt) when the toolchain exposes the builtin; weight streams use
// global_prefetch. Workspace layout (bytes):
//   [0)        wa_t  bf16 [64][1024]      (w_qkv_a^T)
//   [131072)   wb_t  bf16 [3072][64]      (w_qkv_b^T)
//   [524288)   pa_t  bf16 [64][1024]      (w_proj_a^T)
//   [655360)   pb_t  bf16 [1024][64]      (w_proj_b^T)
//   [786432)   q     bf16 [2,16,2048,64]  (pre-scaled by 1/8)
//   [+8MB)     k     bf16 [2,16,2048,64]
//   [+16MB)    vt    bf16 [2,16,64,2048]  (v transposed)
//   [+24MB)    ao    bf16 [4096][1024]    (attention output)
// Total ~33 MB.
// ---------------------------------------------------------------------------

typedef __bf16 bf16_t;
typedef __attribute__((ext_vector_type(16))) __bf16 v16bf;
typedef __attribute__((ext_vector_type(8)))  __bf16 v8bf;
typedef __attribute__((ext_vector_type(8)))  float  v8f;

#define N_  2048
#define C_  1024
#define R_  64
#define H_  16
#define DH  64

#if __has_builtin(__builtin_amdgcn_global_load_async_to_lds_b128)
#define USE_ASYNC_LDS 1
#else
#define USE_ASYNC_LDS 0
#endif

#if USE_ASYNC_LDS
// Signature (from clang diagnostic): (int4 AS(1)* src, int4 AS(3)* dst, imm, imm)
typedef int v4i32 __attribute__((vector_size(16)));
typedef __attribute__((address_space(1))) v4i32 gv4i;
typedef __attribute__((address_space(3))) v4i32 lv4i;

// 16-byte global -> LDS async copy (one lane, one b128 transfer).
__device__ __forceinline__ void async_cp16(const void* g, void* l) {
  __builtin_amdgcn_global_load_async_to_lds_b128(
      (gv4i*)(uintptr_t)g, (lv4i*)(unsigned int)(uintptr_t)l, 0, 0);
}

__device__ __forceinline__ void wait_async0() {
#if __has_builtin(__builtin_amdgcn_s_wait_asynccnt)
  __builtin_amdgcn_s_wait_asynccnt(0);
#else
  asm volatile("s_wait_asynccnt 0x0" ::: "memory");
#endif
}
#endif  // USE_ASYNC_LDS

__device__ __forceinline__ v8f wmma_bf16(v16bf a, v16bf b, v8f c) {
  // (neg_a, A, neg_b, B, c_mod, C, reuse_a, reuse_b)
  return __builtin_amdgcn_wmma_f32_16x16x32_bf16(false, a, false, b, (short)0, c,
                                                 false, false);
}

// A fragment (16x32 bf16, M x K) from a row-major [16][ldk] tile.
// lane l: m = l&15, hi = l>>4; e<8 -> K = k0+8*hi+e ; e>=8 -> K = k0+16+8*hi+(e-8)
__device__ __forceinline__ v16bf load_frag_a(const bf16_t* tile, int ldk, int k0, int lane) {
  const int m = lane & 15, hi = lane >> 4;
  const bf16_t* p = tile + m * ldk + k0 + 8 * hi;
  v8bf lo = *(const v8bf*)(p);
  v8bf hh = *(const v8bf*)(p + 16);
  v16bf r;
#pragma unroll
  for (int i = 0; i < 8; ++i) { r[i] = lo[i]; r[i + 8] = hh[i]; }
  return r;
}

// B fragment (32x16 bf16, K x N) gathered from a row-major B^T tile [16 n][ldk k].
// lane l: n = l&15, hi = l>>4; e<8 -> K = k0+8*hi+e ; e>=8 -> K = k0+16+8*hi+(e-8)
__device__ __forceinline__ v16bf load_frag_b(const bf16_t* tile, int ldk, int k0, int lane) {
  const int n = lane & 15, hi = lane >> 4;
  const bf16_t* p = tile + n * ldk + k0 + 8 * hi;
  v8bf lo = *(const v8bf*)(p);
  v8bf hh = *(const v8bf*)(p + 16);
  v16bf r;
#pragma unroll
  for (int i = 0; i < 8; ++i) { r[i] = lo[i]; r[i + 8] = hh[i]; }
  return r;
}

__device__ __forceinline__ float gelu_exact(float x) {
  return 0.5f * x * (1.0f + erff(x * 0.70710678118654752f));
}

// ---------------------------------------------------------------------------
// Kernel 0: cast + transpose LoRA weights to bf16 [n][k] layouts.
// ---------------------------------------------------------------------------
__global__ __launch_bounds__(256)
void prep_weights_kernel(const float* __restrict__ wqa, const float* __restrict__ wqb,
                         const float* __restrict__ wpa, const float* __restrict__ wpb,
                         bf16_t* __restrict__ wa_t, bf16_t* __restrict__ wb_t,
                         bf16_t* __restrict__ pa_t, bf16_t* __restrict__ pb_t) {
  const int idx = blockIdx.x * 256 + threadIdx.x;
  if (idx < R_ * C_) {                 // [64][1024] <- [1024][64]
    const int r = idx / C_, c = idx % C_;
    wa_t[idx] = (bf16_t)wqa[c * R_ + r];
    pa_t[idx] = (bf16_t)wpa[c * R_ + r];
  }
  if (idx < 3 * C_ * R_) {             // [3072][64] <- [64][3072]
    const int n = idx / R_, r = idx % R_;
    wb_t[idx] = (bf16_t)wqb[r * (3 * C_) + n];
  }
  if (idx < C_ * R_) {                 // [1024][64] <- [64][1024]
    const int n = idx / R_, r = idx % R_;
    pb_t[idx] = (bf16_t)wpb[r * C_ + n];
  }
}

// ---------------------------------------------------------------------------
// Kernel 1: qkv = gelu(x @ Wa) @ Wb ; scatter q (scaled), k, v^T as bf16.
// 256 threads (8 waves), 128 rows per WG; blockIdx.y splits stage-2 columns
// 4 ways (stage 1 is rank-64 and cheap to recompute).
// ---------------------------------------------------------------------------
__global__ __launch_bounds__(256)
void lora_qkv_kernel(const float* __restrict__ x,
                     const bf16_t* __restrict__ wa_t,   // [64][1024]
                     const bf16_t* __restrict__ wb_t,   // [3072][64]
                     bf16_t* __restrict__ qb, bf16_t* __restrict__ kb,
                     bf16_t* __restrict__ vtb) {
  __shared__ __align__(16) bf16_t xs[128 * 32];   // x K-chunk, bf16
  __shared__ __align__(16) bf16_t Ts[128 * 64];   // T = gelu(x@Wa), bf16
  const int t = threadIdx.x;
  const int lane = t & 31, w = t >> 5;
  const int nl = lane & 15, hi = lane >> 4;
  const int m0 = blockIdx.x * 128;

  // ---- stage 1: T = gelu(x @ Wa)  (K = 1024, chunks of 32) ----
  v8f acc[4] = {};
  for (int kk = 0; kk < C_; kk += 32) {
    {  // stage 128x32 f32 chunk of x into LDS as bf16
      const int r = t >> 1, c0 = (t & 1) * 16;
      const float* src = x + (size_t)(m0 + r) * C_ + kk + c0;
#pragma unroll
      for (int i = 0; i < 4; ++i) {
        float4 v = *(const float4*)(src + i * 4);
        bf16_t* dst = &xs[r * 32 + c0 + i * 4];
        dst[0] = (bf16_t)v.x; dst[1] = (bf16_t)v.y;
        dst[2] = (bf16_t)v.z; dst[3] = (bf16_t)v.w;
      }
    }
    __builtin_prefetch(wa_t + kk + 32, 0, 1);  // next Wa K-chunk -> cache
    __syncthreads();
    v16bf a = load_frag_a(xs + 16 * w * 32, 32, 0, lane);
#pragma unroll
    for (int ct = 0; ct < 4; ++ct) {
      v16bf b = load_frag_b(wa_t + ct * 16 * C_, C_, kk, lane);  // global B frag
      acc[ct] = wmma_bf16(a, b, acc[ct]);
    }
    __syncthreads();
  }
#pragma unroll
  for (int ct = 0; ct < 4; ++ct)
#pragma unroll
    for (int j = 0; j < 8; ++j)
      Ts[(16 * w + j + 8 * hi) * 64 + ct * 16 + nl] = (bf16_t)gelu_exact(acc[ct][j]);
  __syncthreads();

  // ---- stage 2: qkv = T @ Wb  (K = 64; this block does 48 of 192 col tiles) ----
  v16bf a0 = load_frag_a(Ts + 16 * w * 64, 64, 0, lane);
  v16bf a1 = load_frag_a(Ts + 16 * w * 64, 64, 32, lane);
  const int ct0 = blockIdx.y * 48;
  for (int ct = ct0; ct < ct0 + 48; ++ct) {
    __builtin_prefetch(wb_t + (ct + 4) * 16 * 64, 0, 1);
    v16bf b0 = load_frag_b(wb_t + ct * 16 * 64, 64, 0, lane);
    v16bf b1 = load_frag_b(wb_t + ct * 16 * 64, 64, 32, lane);
    v8f c = {};
    c = wmma_bf16(a0, b0, c);
    c = wmma_bf16(a1, b1, c);
    const int c3 = ct * 16;
    const int which = c3 >> 10;           // 0=q 1=k 2=v
    const int rem = c3 & 1023;
    const int h = rem >> 6;
    const int d = (rem & 63) + nl;
#pragma unroll
    for (int j = 0; j < 8; ++j) {
      const int gr = m0 + 16 * w + j + 8 * hi;
      const int bbi = gr >> 11, nn = gr & (N_ - 1);
      const size_t bh = (size_t)(bbi * H_ + h);
      const float v = c[j];
      if (which == 0)      qb [(bh * N_ + nn) * DH + d] = (bf16_t)(v * 0.125f);  // Dh^-0.5
      else if (which == 1) kb [(bh * N_ + nn) * DH + d] = (bf16_t)v;
      else                 vtb[(bh * DH + d) * N_ + nn] = (bf16_t)v;             // transposed
    }
  }
}

// ---------------------------------------------------------------------------
// Kernel 2: flash attention. One WG per (b, h, 128 q-rows); 8 waves x 16 rows.
// K/V blocks double-buffered in LDS; async global->LDS copies overlap the
// previous block's WMMA work when available.
// ---------------------------------------------------------------------------
__global__ __launch_bounds__(256)
void attn_kernel(const bf16_t* __restrict__ qb, const bf16_t* __restrict__ kb,
                 const bf16_t* __restrict__ vtb, bf16_t* __restrict__ ao) {
  __shared__ __align__(16) bf16_t qs[128 * 64];        // 16 KB
  __shared__ __align__(16) bf16_t ks[2][64 * 64];      // 16 KB  [n_kv][d] x2
  __shared__ __align__(16) bf16_t vs[2][64 * 64];      // 16 KB  [d][n_kv] x2
  __shared__ __align__(16) bf16_t ps[8][16 * 64];      // 16 KB wave-private P
  const int t = threadIdx.x;
  const int lane = t & 31, w = t >> 5;
  const int nl = lane & 15, hi = lane >> 4;
  const int qt = blockIdx.x & 15;
  const int h  = (blockIdx.x >> 4) & 15;
  const int bbi = blockIdx.x >> 8;
  const size_t bh = (size_t)(bbi * H_ + h);
  const bf16_t* qg  = qb  + (bh * N_ + (size_t)qt * 128) * DH;
  const bf16_t* kg  = kb  + bh * N_ * DH;
  const bf16_t* vtg = vtb + bh * DH * N_;

  // Issue (or perform) the K / V^T copies for block kv0 into buffer `buf`.
  auto issue_kv = [&](int kv0, int buf) {
#if USE_ASYNC_LDS
    const bf16_t* ksrc = kg + (size_t)kv0 * DH;          // contiguous [64][64]
    async_cp16(ksrc + (size_t)t * 8,         &ks[buf][t * 8]);
    async_cp16(ksrc + (size_t)(t + 256) * 8, &ks[buf][(t + 256) * 8]);
#pragma unroll
    for (int rr = 0; rr < 2; ++rr) {                     // v^T: 64 rows, stride N_
      const int i = t + rr * 256;
      const int d = i >> 3, c8 = (i & 7) * 8;
      async_cp16(vtg + (size_t)d * N_ + kv0 + c8, &vs[buf][d * 64 + c8]);
    }
#else
    {
      const uint4* src = (const uint4*)(kg + (size_t)kv0 * DH);
      uint4* dst = (uint4*)ks[buf];
      dst[t] = src[t];
      dst[t + 256] = src[t + 256];
    }
#pragma unroll
    for (int rr = 0; rr < 2; ++rr) {
      const int i = t + rr * 256;
      const int d = i >> 3, c8 = (i & 7) * 8;
      *(uint4*)&vs[buf][d * 64 + c8] = *(const uint4*)(vtg + (size_t)d * N_ + kv0 + c8);
    }
#endif
  };

  {  // q tile: contiguous 8192 bf16 = 1024 uint4
    const uint4* src = (const uint4*)qg;
    uint4* dst = (uint4*)qs;
#pragma unroll
    for (int i = 0; i < 4; ++i) dst[t + i * 256] = src[t + i * 256];
  }
  issue_kv(0, 0);  // prologue: block 0 -> buffer 0 (overlaps nothing yet)
  __syncthreads();

  v16bf qa0 = load_frag_a(qs + 16 * w * 64, 64, 0, lane);
  v16bf qa1 = load_frag_a(qs + 16 * w * 64, 64, 32, lane);

  v8f o[4] = {};
  float m_i[8], l_i[8];
#pragma unroll
  for (int j = 0; j < 8; ++j) { m_i[j] = -INFINITY; l_i[j] = 0.0f; }
  bf16_t* pw = ps[w];

  for (int it = 0; it < N_ / 64; ++it) {
#if USE_ASYNC_LDS
    wait_async0();               // my async copies for block `it` are done
#endif
    __syncthreads();             // everyone's copies done; prev buffer free
    if (it + 1 < N_ / 64) issue_kv((it + 1) * 64, (it + 1) & 1);
    const bf16_t* kcur = ks[it & 1];
    const bf16_t* vcur = vs[it & 1];

    // ---- scores S = q @ k^T (4 column tiles of 16, K = 64) ----
    v8f s[4];
#pragma unroll
    for (int ct = 0; ct < 4; ++ct) {
      v16bf b0 = load_frag_b(kcur + ct * 16 * 64, 64, 0, lane);
      v16bf b1 = load_frag_b(kcur + ct * 16 * 64, 64, 32, lane);
      v8f c = {};
      c = wmma_bf16(qa0, b0, c);
      c = wmma_bf16(qa1, b1, c);
      s[ct] = c;
    }

    // ---- online softmax (row = j + 8*hi, spread over 16 adjacent lanes) ----
#pragma unroll
    for (int j = 0; j < 8; ++j) {
      float r = fmaxf(fmaxf(s[0][j], s[1][j]), fmaxf(s[2][j], s[3][j]));
#pragma unroll
      for (int msk = 8; msk >= 1; msk >>= 1) r = fmaxf(r, __shfl_xor(r, msk, 32));
      const float mn = fmaxf(m_i[j], r);
      float sum = 0.0f;
#pragma unroll
      for (int ct = 0; ct < 4; ++ct) {
        const float p = __expf(s[ct][j] - mn);
        s[ct][j] = p;
        sum += p;
      }
#pragma unroll
      for (int msk = 8; msk >= 1; msk >>= 1) sum += __shfl_xor(sum, msk, 32);
      const float sc = __expf(m_i[j] - mn);
      l_i[j] = l_i[j] * sc + sum;
      m_i[j] = mn;
#pragma unroll
      for (int ct = 0; ct < 4; ++ct) o[ct][j] *= sc;
    }

    // ---- P (bf16) -> wave-private LDS, re-read as A fragment ----
#pragma unroll
    for (int ct = 0; ct < 4; ++ct)
#pragma unroll
      for (int j = 0; j < 8; ++j)
        pw[(j + 8 * hi) * 64 + ct * 16 + nl] = (bf16_t)s[ct][j];

    v16bf pa0 = load_frag_a(pw, 64, 0, lane);
    v16bf pa1 = load_frag_a(pw, 64, 32, lane);
#pragma unroll
    for (int dt = 0; dt < 4; ++dt) {
      v16bf b0 = load_frag_b(vcur + dt * 16 * 64, 64, 0, lane);
      v16bf b1 = load_frag_b(vcur + dt * 16 * 64, 64, 32, lane);
      o[dt] = wmma_bf16(pa0, b0, o[dt]);
      o[dt] = wmma_bf16(pa1, b1, o[dt]);
    }
    // no trailing barrier: the barrier at the top of the next iteration
    // protects the buffer being refilled.
  }

  // ---- finalize: o /= l ; write [B,N,H*Dh] bf16 ----
#pragma unroll
  for (int dt = 0; dt < 4; ++dt)
#pragma unroll
    for (int j = 0; j < 8; ++j) {
      const int nn = qt * 128 + 16 * w + j + 8 * hi;
      ao[((size_t)bbi * N_ + nn) * C_ + h * DH + dt * 16 + nl] =
          (bf16_t)(o[dt][j] / l_i[j]);
    }
}

// ---------------------------------------------------------------------------
// Kernel 3: out = gelu(ao @ Pa + ba) @ Pb + bb   (f32 output)
// blockIdx.y splits stage-2 columns 2 ways.
// ---------------------------------------------------------------------------
__global__ __launch_bounds__(256)
void lora_proj_kernel(const bf16_t* __restrict__ ao,
                      const bf16_t* __restrict__ pa_t,  // [64][1024]
                      const float*  __restrict__ bpa,   // [64]
                      const bf16_t* __restrict__ pb_t,  // [1024][64]
                      const float*  __restrict__ bpb,   // [1024]
                      float* __restrict__ out) {
  __shared__ __align__(16) bf16_t Us[128 * 64];
  const int t = threadIdx.x;
  const int lane = t & 31, w = t >> 5;
  const int nl = lane & 15, hi = lane >> 4;
  const int m0 = blockIdx.x * 128;

  // ---- stage 1: U = gelu(ao @ Pa + ba)  (A fragments straight from global) ----
  v8f acc[4] = {};
  const bf16_t* arow = ao + (size_t)(m0 + 16 * w) * C_;
  for (int kk = 0; kk < C_; kk += 32) {
    __builtin_prefetch(pa_t + kk + 32, 0, 1);
    v16bf a = load_frag_a(arow, C_, kk, lane);
#pragma unroll
    for (int ct = 0; ct < 4; ++ct) {
      v16bf b = load_frag_b(pa_t + ct * 16 * C_, C_, kk, lane);
      acc[ct] = wmma_bf16(a, b, acc[ct]);
    }
  }
#pragma unroll
  for (int ct = 0; ct < 4; ++ct)
#pragma unroll
    for (int j = 0; j < 8; ++j) {
      const float u = gelu_exact(acc[ct][j] + bpa[ct * 16 + nl]);
      Us[(16 * w + j + 8 * hi) * 64 + ct * 16 + nl] = (bf16_t)u;
    }
  __syncthreads();

  // ---- stage 2: out = U @ Pb + bb (this block: 32 of 64 col tiles) ----
  v16bf a0 = load_frag_a(Us + 16 * w * 64, 64, 0, lane);
  v16bf a1 = load_frag_a(Us + 16 * w * 64, 64, 32, lane);
  const int ct0 = blockIdx.y * 32;
  for (int ct = ct0; ct < ct0 + 32; ++ct) {
    __builtin_prefetch(pb_t + (ct + 4) * 16 * 64, 0, 1);
    v16bf b0 = load_frag_b(pb_t + ct * 16 * 64, 64, 0, lane);
    v16bf b1 = load_frag_b(pb_t + ct * 16 * 64, 64, 32, lane);
    v8f c = {};
    c = wmma_bf16(a0, b0, c);
    c = wmma_bf16(a1, b1, c);
    const float bias = bpb[ct * 16 + nl];
#pragma unroll
    for (int j = 0; j < 8; ++j)
      out[(size_t)(m0 + 16 * w + j + 8 * hi) * C_ + ct * 16 + nl] = c[j] + bias;
  }
}

// ---------------------------------------------------------------------------
extern "C" void kernel_launch(void* const* d_in, const int* in_sizes, int n_in,
                              void* d_out, int out_size, void* d_ws, size_t ws_size,
                              hipStream_t stream) {
  const float* x   = (const float*)d_in[0];
  const float* wqa = (const float*)d_in[1];
  const float* wqb = (const float*)d_in[2];
  const float* wpa = (const float*)d_in[3];
  const float* bpa = (const float*)d_in[4];
  const float* wpb = (const float*)d_in[5];
  const float* bpb = (const float*)d_in[6];
  float* out = (float*)d_out;

  char* ws = (char*)d_ws;
  bf16_t* wa_t = (bf16_t*)(ws + 0);
  bf16_t* wb_t = (bf16_t*)(ws + 131072);
  bf16_t* pa_t = (bf16_t*)(ws + 524288);
  bf16_t* pb_t = (bf16_t*)(ws + 655360);
  bf16_t* qb   = (bf16_t*)(ws + 786432);
  bf16_t* kb   = (bf16_t*)(ws + 786432 + 1 * 8388608);
  bf16_t* vtb  = (bf16_t*)(ws + 786432 + 2 * 8388608);
  bf16_t* ao   = (bf16_t*)(ws + 786432 + 3 * 8388608);

  prep_weights_kernel<<<768, 256, 0, stream>>>(wqa, wqb, wpa, wpb,
                                               wa_t, wb_t, pa_t, pb_t);
  lora_qkv_kernel<<<dim3(32, 4), 256, 0, stream>>>(x, wa_t, wb_t, qb, kb, vtb);
  attn_kernel<<<512, 256, 0, stream>>>(qb, kb, vtb, ao);
  lora_proj_kernel<<<dim3(32, 2), 256, 0, stream>>>(ao, pa_t, bpa, pb_t, bpb, out);
}
